// TemporalGraphBlock_16054587752810
// MI455X (gfx1250) — matrix operand order repaired and checked by reading, not compiled, not run on hardware
//
#include <hip/hip_runtime.h>
#include <hip/hip_bf16.h>
#include <math.h>

typedef __attribute__((ext_vector_type(16))) __bf16 v16bf;
typedef __attribute__((ext_vector_type(8)))  float  v8f;

#define B_   8
#define T_   4096
#define D_   256
#define H_   4
#define C_   64
#define E_   65536
#define ETOT (E_ + T_)
#define BT   (B_ * T_)
#define FF_  (4 * D_)

#define BS_STRIDE 36          // LDS row stride (elem): 72B = 18 banks, conflict-light
#define BS_BUF    (64 * BS_STRIDE)

// ---------------------------------------------------------------- utilities

static __device__ __forceinline__ float gelu_exact(float x) {
    return 0.5f * x * (1.0f + erff(x * 0.70710678118654752440f));
}

static __device__ __forceinline__ void atomicMaxF(float* addr, float val) {
    unsigned int* ua = reinterpret_cast<unsigned int*>(addr);
    unsigned int cur = *ua;
    while (__uint_as_float(cur) < val) {
        unsigned int prev = atomicCAS(ua, cur, __float_as_uint(val));
        if (prev == cur) break;
        cur = prev;
    }
}

// ------------------------------------------------- weight convert+transpose
// W[K,N] fp32 row-major  ->  Wt[N,K] bf16 row-major (so each WMMA B-fragment
// lane reads 16 contiguous bf16 = 32 bytes).
__global__ void wt_transpose_bf16(const float* __restrict__ w,
                                  __hip_bfloat16* __restrict__ wt,
                                  int K, int N) {
    int idx = blockIdx.x * blockDim.x + threadIdx.x;
    if (idx >= K * N) return;
    int k = idx / N, n = idx % N;
    wt[(size_t)n * K + k] = __float2bfloat16(w[idx]);
}

// ------------------------------------------------------------ layernorm (D=256)
// one wave per row: 32 lanes x 8 strided elements, shfl_xor tree reduction.
__global__ void ln_bf16_kernel(const float* __restrict__ x,
                               const float* __restrict__ g,
                               const float* __restrict__ b,
                               __hip_bfloat16* __restrict__ out, int rows) {
    int wave = threadIdx.x >> 5, lane = threadIdx.x & 31;
    int row = blockIdx.x * (blockDim.x >> 5) + wave;
    if (row >= rows) return;
    const float* xr = x + (size_t)row * D_;
    float v[8], s = 0.f, ss = 0.f;
#pragma unroll
    for (int i = 0; i < 8; ++i) {
        v[i] = xr[lane + i * 32];
        s += v[i]; ss += v[i] * v[i];
    }
#pragma unroll
    for (int o = 16; o > 0; o >>= 1) {
        s  += __shfl_xor(s,  o, 32);
        ss += __shfl_xor(ss, o, 32);
    }
    float mu   = s  * (1.0f / D_);
    float var  = ss * (1.0f / D_) - mu * mu;
    float rstd = rsqrtf(var + 1e-5f);
    __hip_bfloat16* orow = out + (size_t)row * D_;
#pragma unroll
    for (int i = 0; i < 8; ++i) {
        int c = lane + i * 32;
        orow[c] = __float2bfloat16((v[i] - mu) * rstd * g[c] + b[c]);
    }
}

// --------------------------------------------------------------- WMMA GEMM
// C[M,N] = A[M,K](bf16,row) @ Wt[N,K](bf16,row)  via v_wmma_f32_16x16x32_bf16
// block = 128 threads (4 waves) computes a 64x64 tile: wave w owns rows
// m0+16w..+16; the 64x32 B tile is double-buffered in LDS, one barrier/k-step.
// All 4 B fragments are preloaded before the 4 WMMAs so a single dscnt wait
// covers the whole matrix chain.
// EPI: 0 = plain f32 store, 1 = +bias,GELU -> bf16, 2 = +bias,+residual -> f32
union Frag { v16bf v; uint4 q[2]; };

template <int EPI>
__global__ void gemm_wmma_bf16(const __hip_bfloat16* __restrict__ A,
                               const __hip_bfloat16* __restrict__ Wt,
                               const float* __restrict__ bias,
                               const float* __restrict__ res,
                               float* __restrict__ outF,
                               __hip_bfloat16* __restrict__ outBF,
                               int M, int N, int K) {
    __shared__ __align__(16) __hip_bfloat16 Bs[2 * BS_BUF];

    const int tid  = threadIdx.x;
    const int wave = tid >> 5;
    const int lane = tid & 31;
    const int lhi  = lane >> 4;   // K-half selector (contiguous k = 16*lhi + s in A/B)
    const int lLo  = lane & 15;   // M row (A) / N col (B,C,D)
    const int m0 = blockIdx.x * 64 + wave * 16;
    const int n0 = blockIdx.y * 64;

    v8f zero = {};
    v8f acc[4];
#pragma unroll
    for (int t = 0; t < 4; ++t) acc[t] = zero;

    const __hip_bfloat16* aRow = A + (size_t)(m0 + lLo) * K;

    // cooperative B staging: thread -> row nl (0..63), k-chunk kc (0 or 16)
    const int nl = tid >> 1;
    const int kc = (tid & 1) * 16;
    const __hip_bfloat16* bRow = Wt + (size_t)(blockIdx.y * 64 + nl) * K + kc;
    const int bsOff = nl * BS_STRIDE + kc;

    const int kIters = K >> 5;

    // prologue: stage k-step 0 into buffer 0
    {
        uint4 b0 = *reinterpret_cast<const uint4*>(bRow);
        uint4 b1 = *reinterpret_cast<const uint4*>(bRow + 8);
        *reinterpret_cast<uint4*>(&Bs[bsOff])     = b0;
        *reinterpret_cast<uint4*>(&Bs[bsOff + 8]) = b1;
    }
    __syncthreads();

    for (int j = 0; j < kIters; ++j) {
        const int k0 = j << 5;
        const __hip_bfloat16* buf = &Bs[(j & 1) * BS_BUF];
        const bool more = (j + 1 < kIters);

        // issue next tile's global loads early (consumed at end of iteration)
        uint4 nb0, nb1;
        if (more) {
            nb0 = *reinterpret_cast<const uint4*>(bRow + k0 + 32);
            nb1 = *reinterpret_cast<const uint4*>(bRow + k0 + 40);
            __builtin_prefetch(aRow + k0 + 64, 0, 3);
            __builtin_prefetch(bRow + k0 + 64, 0, 3);
        }

        // A fragment 16x32: frag[0..15] = A[m][k0+16*lhi .. +16)
        Frag a;
        const __hip_bfloat16* ap = aRow + k0 + lhi * 16;
        a.q[0] = *reinterpret_cast<const uint4*>(ap);
        a.q[1] = *reinterpret_cast<const uint4*>(ap + 8);

        // preload ALL B fragments (32x16 each) from LDS, then issue the WMMAs
        Frag bf[4];
#pragma unroll
        for (int t = 0; t < 4; ++t) {
            const __hip_bfloat16* bp = &buf[(t * 16 + lLo) * BS_STRIDE + lhi * 16];
            bf[t].q[0] = *reinterpret_cast<const uint4*>(bp);
            bf[t].q[1] = *reinterpret_cast<const uint4*>(bp + 8);
        }
#pragma unroll
        for (int t = 0; t < 4; ++t) {
            acc[t] = __builtin_amdgcn_wmma_f32_16x16x32_bf16(
                false, a.v, false, bf[t].v, (short)0, acc[t], false, false);
        }

        // stage next tile into the alternate buffer; writes race with nothing:
        // buffer (j+1)&1 was last read in iteration j-1, completed before the
        // barrier that ended that iteration.
        if (more) {
            __hip_bfloat16* dst = &Bs[((j + 1) & 1) * BS_BUF + bsOff];
            *reinterpret_cast<uint4*>(dst)     = nb0;
            *reinterpret_cast<uint4*>(dst + 8) = nb1;
        }
        __syncthreads();
    }

#pragma unroll
    for (int t = 0; t < 4; ++t) {
        int col = n0 + t * 16 + lLo;
        float bv = (EPI != 0) ? bias[col] : 0.0f;
#pragma unroll
        for (int r = 0; r < 8; ++r) {
            int row = m0 + r + lhi * 8;       // ISA f32 C/D layout
            float vv = acc[t][r] + bv;
            if (EPI == 1) {
                outBF[(size_t)row * N + col] = __float2bfloat16(gelu_exact(vv));
            } else if (EPI == 2) {
                outF[(size_t)row * N + col] = vv + res[(size_t)row * N + col];
            } else {
                outF[(size_t)row * N + col] = vv;
            }
        }
    }
}

// --------------------------------------------------------- GAT edge pipeline

// a_s[b,t,h] = sum_c xp[b,t,h,c]*att_src[h,c]; a_d likewise
__global__ void attn_coef_kernel(const float* __restrict__ xp,
                                 const float* __restrict__ att_src,
                                 const float* __restrict__ att_dst,
                                 float* __restrict__ a_s,
                                 float* __restrict__ a_d) {
    int idx = blockIdx.x * blockDim.x + threadIdx.x;   // over BT*H
    if (idx >= BT * H_) return;
    int h = idx & (H_ - 1);
    int bt = idx >> 2;
    const float* xr = xp + (size_t)bt * D_ + h * C_;
    const float* as = att_src + h * C_;
    const float* ad = att_dst + h * C_;
    float s = 0.f, d = 0.f;
#pragma unroll 8
    for (int c = 0; c < C_; ++c) { float v = xr[c]; s += v * as[c]; d += v * ad[c]; }
    a_s[idx] = s; a_d[idx] = d;
}

__global__ void init_kernel(float* __restrict__ y,
                            float* __restrict__ mmax,
                            float* __restrict__ denom) {
    int idx = blockIdx.x * blockDim.x + threadIdx.x;
    if (idx < BT * D_) y[idx] = 0.0f;
    if (idx < BT * H_) { mmax[idx] = -3.402823466e38f; denom[idx] = 0.0f; }
}

// leaky_relu(a_s[src]+a_d[dst]) + segment max over dst (self loops appended)
__global__ void edge_score_kernel(const int* __restrict__ ei,
                                  const float* __restrict__ a_s,
                                  const float* __restrict__ a_d,
                                  float* __restrict__ escore,
                                  float* __restrict__ mmax) {
    int idx = blockIdx.x * blockDim.x + threadIdx.x;   // over B*ETOT*H
    if (idx >= B_ * ETOT * H_) return;
    int h = idx & 3;
    int rem = idx >> 2;
    int e = rem % ETOT, b = rem / ETOT;
    int s = (e < E_) ? ei[e]      : (e - E_);
    int d = (e < E_) ? ei[E_ + e] : (e - E_);
    float sc = a_s[(b * T_ + s) * H_ + h] + a_d[(b * T_ + d) * H_ + h];
    sc = (sc > 0.0f) ? sc : 0.2f * sc;
    escore[idx] = sc;
    atomicMaxF(&mmax[(b * T_ + d) * H_ + h], sc);
}

__global__ void edge_exp_kernel(const int* __restrict__ ei,
                                float* __restrict__ escore,
                                const float* __restrict__ mmax,
                                float* __restrict__ denom) {
    int idx = blockIdx.x * blockDim.x + threadIdx.x;
    if (idx >= B_ * ETOT * H_) return;
    int h = idx & 3;
    int rem = idx >> 2;
    int e = rem % ETOT, b = rem / ETOT;
    int d = (e < E_) ? ei[E_ + e] : (e - E_);
    float ex = expf(escore[idx] - mmax[(b * T_ + d) * H_ + h]);
    escore[idx] = ex;
    atomicAdd(&denom[(b * T_ + d) * H_ + h], ex);
}

// y[b,dst,:] += alpha * xp[b,src,:]   (block = 1 edge, 256 threads = channels)
__global__ void edge_scatter_kernel(const int* __restrict__ ei,
                                    const float* __restrict__ escore,
                                    const float* __restrict__ denom,
                                    const float* __restrict__ xp,
                                    float* __restrict__ y) {
    int be = blockIdx.x;           // b*ETOT + e
    int c  = threadIdx.x;          // 0..255
    int e = be % ETOT, b = be / ETOT;
    int s = (e < E_) ? ei[e]      : (e - E_);
    int d = (e < E_) ? ei[E_ + e] : (e - E_);
    int h = c >> 6;
    float alpha = escore[(size_t)be * H_ + h] / denom[(b * T_ + d) * H_ + h];
    atomicAdd(&y[(size_t)(b * T_ + d) * D_ + c],
              alpha * xp[(size_t)(b * T_ + s) * D_ + c]);
}

// xn = x + y + gat_bias
__global__ void residual1_kernel(const float* __restrict__ x,
                                 const float* __restrict__ y,
                                 const float* __restrict__ gbias,
                                 float* __restrict__ xn) {
    int idx = blockIdx.x * blockDim.x + threadIdx.x;
    if (idx >= BT * D_) return;
    int c = idx & (D_ - 1);
    xn[idx] = x[idx] + y[idx] + gbias[c];
}

// ------------------------------------------------------------------ launch

extern "C" void kernel_launch(void* const* d_in, const int* in_sizes, int n_in,
                              void* d_out, int out_size, void* d_ws, size_t ws_size,
                              hipStream_t stream) {
    const float* x        = (const float*)d_in[0];
    /* d_in[1] adjacency: unused by reference math */
    const int*   edge_ind = (const int*)  d_in[2];
    const float* gat_w    = (const float*)d_in[3];
    const float* att_src  = (const float*)d_in[4];
    const float* att_dst  = (const float*)d_in[5];
    const float* gat_bias = (const float*)d_in[6];
    const float* ln1_g    = (const float*)d_in[7];
    const float* ln1_b    = (const float*)d_in[8];
    const float* ln2_g    = (const float*)d_in[9];
    const float* ln2_b    = (const float*)d_in[10];
    const float* ff_w1    = (const float*)d_in[11];
    const float* ff_b1    = (const float*)d_in[12];
    const float* ff_w2    = (const float*)d_in[13];
    const float* ff_b2    = (const float*)d_in[14];

    char* ws = (char*)d_ws;
    size_t off = 0;
    auto alloc = [&](size_t bytes) -> void* {
        void* p = ws + off;
        off += (bytes + 255) & ~(size_t)255;
        return p;
    };

    __hip_bfloat16* hBF   = (__hip_bfloat16*)alloc((size_t)BT * D_ * 2);   // LN1/LN2 out (reused)
    float*          xp    = (float*)         alloc((size_t)BT * D_ * 4);   // GAT projection
    float*          a_s   = (float*)         alloc((size_t)BT * H_ * 4);
    float*          a_d   = (float*)         alloc((size_t)BT * H_ * 4);
    float*          esc   = (float*)         alloc((size_t)B_ * ETOT * H_ * 4);
    float*          mmax  = (float*)         alloc((size_t)BT * H_ * 4);
    float*          denom = (float*)         alloc((size_t)BT * H_ * 4);
    float*          yacc  = (float*)         alloc((size_t)BT * D_ * 4);
    float*          xn    = (float*)         alloc((size_t)BT * D_ * 4);
    __hip_bfloat16* act   = (__hip_bfloat16*)alloc((size_t)BT * FF_ * 2);
    __hip_bfloat16* gatWt = (__hip_bfloat16*)alloc((size_t)D_ * D_ * 2);
    __hip_bfloat16* w1t   = (__hip_bfloat16*)alloc((size_t)D_ * FF_ * 2);
    __hip_bfloat16* w2t   = (__hip_bfloat16*)alloc((size_t)FF_ * D_ * 2);
    (void)ws_size; (void)in_sizes; (void)n_in; (void)out_size;

    // --- weight prep (bf16, transposed to [N,K])
    wt_transpose_bf16<<<(D_ * D_  + 255) / 256, 256, 0, stream>>>(gat_w, gatWt, D_,  D_);
    wt_transpose_bf16<<<(D_ * FF_ + 255) / 256, 256, 0, stream>>>(ff_w1, w1t,   D_,  FF_);
    wt_transpose_bf16<<<(FF_ * D_ + 255) / 256, 256, 0, stream>>>(ff_w2, w2t,   FF_, D_);

    // --- LN1 -> h (bf16)
    ln_bf16_kernel<<<BT / 8, 256, 0, stream>>>(x, ln1_g, ln1_b, hBF, BT);

    // --- GAT projection: xp = h @ gat_w   [32768,256]x[256,256]
    gemm_wmma_bf16<0><<<dim3(BT / 64, D_ / 64), 128, 0, stream>>>(
        hBF, gatWt, nullptr, nullptr, xp, nullptr, BT, D_, D_);

    // --- attention coefficients
    attn_coef_kernel<<<(BT * H_ + 255) / 256, 256, 0, stream>>>(xp, att_src, att_dst, a_s, a_d);

    // --- scatter softmax over edges (+self loops)
    init_kernel<<<(BT * D_ + 255) / 256, 256, 0, stream>>>(yacc, mmax, denom);
    edge_score_kernel<<<(B_ * ETOT * H_ + 255) / 256, 256, 0, stream>>>(edge_ind, a_s, a_d, esc, mmax);
    edge_exp_kernel  <<<(B_ * ETOT * H_ + 255) / 256, 256, 0, stream>>>(edge_ind, esc, mmax, denom);
    edge_scatter_kernel<<<B_ * ETOT, 256, 0, stream>>>(edge_ind, esc, denom, xp, yacc);

    // --- residual 1 (+gat_bias), LN2 -> h2 (bf16, reuse hBF)
    residual1_kernel<<<(BT * D_ + 255) / 256, 256, 0, stream>>>(x, yacc, gat_bias, xn);
    ln_bf16_kernel<<<BT / 8, 256, 0, stream>>>(xn, ln2_g, ln2_b, hBF, BT);

    // --- FFN GEMM1: act = gelu(h2 @ ff_w1 + b1)  [32768,256]x[256,1024] -> bf16
    gemm_wmma_bf16<1><<<dim3(BT / 64, FF_ / 64), 128, 0, stream>>>(
        hBF, w1t, ff_b1, nullptr, nullptr, act, BT, FF_, D_);

    // --- FFN GEMM2 + residual: out = xn + act @ ff_w2 + b2  [32768,1024]x[1024,256]
    gemm_wmma_bf16<2><<<dim3(BT / 64, D_ / 64), 128, 0, stream>>>(
        act, w2t, ff_b2, xn, (float*)d_out, nullptr, BT, D_, FF_);
}
